// ColPaliAttention_29394756174290
// MI455X (gfx1250) — matrix operand order repaired and checked by reading, not compiled, not run on hardware
//
#include <hip/hip_runtime.h>
#include <hip/hip_bf16.h>
#include <stdint.h>

typedef __attribute__((ext_vector_type(16))) __bf16 v16bf;
typedef __attribute__((ext_vector_type(8)))  float  v8f;
typedef uint16_t u16;
typedef uint32_t u32;

#define DEV static __device__ __forceinline__

DEV u16 f2bf(float f) {
  u32 u = __builtin_bit_cast(u32, f);
  u32 r = u + 0x7FFFu + ((u >> 16) & 1u);
  return (u16)(r >> 16);
}
DEV float bf2f(u16 h) { u32 u = (u32)h << 16; return __builtin_bit_cast(float, u); }

union FragU { uint4 q[2]; v16bf v; };

// A-matrix fragment: 16(M) x 32(K) bf16, source row-major [rows][ld].
// Lane l (hf = l>>4, r = l&15) holds row (row0+r), K = {hf*8+0..7, 16+hf*8+0..7}.
DEV v16bf load_a(const u16* base, int ld, int row0, int k0, int lane) {
  const int hf = lane >> 4, r = lane & 15;
  const u16* p = base + (size_t)(row0 + r) * ld + k0 + hf * 8;
  FragU u;
  u.q[0] = *(const uint4*)(p);
  u.q[1] = *(const uint4*)(p + 16);
  return u.v;
}
// B-matrix fragment: 32(K) x 16(N) bf16, source is B^T row-major [N][ld].
// Lane l holds column (n0+r), K = {hf*8+0..7, 16+hf*8+0..7}.
DEV v16bf load_b(const u16* base, int ld, int n0, int k0, int lane) {
  const int hf = lane >> 4, r = lane & 15;
  const u16* p = base + (size_t)(n0 + r) * ld + k0 + hf * 8;
  FragU u;
  u.q[0] = *(const uint4*)(p);
  u.q[1] = *(const uint4*)(p + 16);
  return u.v;
}

DEV v8f wmma_bf16(v16bf a, v16bf b, v8f c) {
  // (neg_a, A, neg_b, B, c_mod, C, reuse_a, reuse_b)
  return __builtin_amdgcn_wmma_f32_16x16x32_bf16(false, a, false, b, (short)0, c,
                                                 false, false);
}

// async global->LDS copy, 16B per lane, ASYNCcnt-tracked (CDNA5 path)
DEV void async_b128(u32 lds_addr, const void* gaddr) {
  asm volatile("global_load_async_to_lds_b128 %0, %1, off"
               :: "v"(lds_addr), "v"((unsigned long long)(uintptr_t)gaddr)
               : "memory");
}

// ---------------------------------------------------------------- convert
__global__ __launch_bounds__(256) void cvt_kernel(const float* __restrict__ in,
                                                  u16* __restrict__ out, size_t n) {
  size_t i = (size_t)blockIdx.x * blockDim.x + threadIdx.x;
  size_t stride = (size_t)gridDim.x * blockDim.x;
  for (; i < n; i += stride) out[i] = f2bf(in[i]);
}

// ---------------------------------------------------------------- V transpose
// v: [B, L, 256] bf16 -> vt: [B, 256, L] bf16  (write-coalesced)
__global__ __launch_bounds__(256)
void transpose_v_kernel(const u16* __restrict__ v, u16* __restrict__ vt, size_t total) {
  size_t i = (size_t)blockIdx.x * blockDim.x + threadIdx.x;
  size_t stride = (size_t)gridDim.x * blockDim.x;
  for (; i < total; i += stride) {
    int l = (int)(i & 2047);
    size_t t = i >> 11;
    int d = (int)(t & 255);
    int b = (int)(t >> 8);
    vt[i] = v[((size_t)b * 2048 + l) * 256 + d];
  }
}

// ---------------------------------------------------------------- GEMM C = A * B^T
// A: [M][K] bf16 row-major, Bt: [N][K] bf16 row-major, C: [M][N] (bf16 or f32).
// Block: 256 threads = 8 waves, wave tile 32x32, block tile 64(M) x 128(N).
template <bool OUT_BF16>
__global__ __launch_bounds__(256)
void gemm_kernel(const u16* __restrict__ A, const u16* __restrict__ Bt,
                 void* __restrict__ Cout, int M, int N, int K) {
  const int lane = threadIdx.x & 31;
  const int wave = threadIdx.x >> 5;
  const int row0 = blockIdx.y * 64 + (wave >> 2) * 32;
  const int col0 = blockIdx.x * 128 + (wave & 3) * 32;

  v8f acc00 = {}, acc01 = {}, acc10 = {}, acc11 = {};
  for (int k0 = 0; k0 < K; k0 += 32) {
    v16bf a0 = load_a(A, K, row0,      k0, lane);
    v16bf a1 = load_a(A, K, row0 + 16, k0, lane);
    v16bf b0 = load_b(Bt, K, col0,      k0, lane);
    v16bf b1 = load_b(Bt, K, col0 + 16, k0, lane);
    acc00 = wmma_bf16(a0, b0, acc00);
    acc01 = wmma_bf16(a0, b1, acc01);
    acc10 = wmma_bf16(a1, b0, acc10);
    acc11 = wmma_bf16(a1, b1, acc11);
  }

  const int hf = lane >> 4, r = lane & 15;
  auto store_tile = [&](v8f c, int ro, int co) {
#pragma unroll
    for (int g = 0; g < 8; ++g) {
      size_t idx = (size_t)(ro + g + 8 * hf) * N + co + r;
      if constexpr (OUT_BF16) ((u16*)Cout)[idx] = f2bf(c[g]);
      else                    ((float*)Cout)[idx] = c[g];
    }
  };
  store_tile(acc00, row0, col0);
  store_tile(acc01, row0, col0 + 16);
  store_tile(acc10, row0 + 16, col0);
  store_tile(acc11, row0 + 16, col0 + 16);
}

// ---------------------------------------------------------------- RoPE (in place, bf16)
// x layout: [B, L, Hn, 256]; thread owns pair (d, d+128) -> no races.
__global__ __launch_bounds__(256)
void rope_kernel(u16* __restrict__ x, const float* __restrict__ cosb,
                 const float* __restrict__ sinb, int L, int Hn, size_t total) {
  size_t i = (size_t)blockIdx.x * blockDim.x + threadIdx.x;
  size_t stride = (size_t)gridDim.x * blockDim.x;
  for (; i < total; i += stride) {
    int dh = (int)(i & 127);
    size_t t = i >> 7;
    int h = (int)(t % Hn); t /= Hn;
    int l = (int)(t % L);
    int b = (int)(t / L);
    size_t base = (((size_t)b * L + l) * Hn + h) * 256;
    float c = cosb[l * 256 + dh];
    float s = sinb[l * 256 + dh];
    float x1 = bf2f(x[base + dh]);
    float x2 = bf2f(x[base + dh + 128]);
    x[base + dh]       = f2bf(x1 * c - x2 * s);
    x[base + dh + 128] = f2bf(x2 * c + x1 * s);
  }
}

// ---------------------------------------------------------------- flash attention
// grid: (L/128, B*H). Block 256 thr = 8 waves; wave owns 16 query rows.
// Q: [B, L, H*256] bf16 (post-RoPE); K: [B, L, 256] bf16; VT: [B, 256, L] bf16;
// mask f32 [B,1,1,L,L]; O: [B, L, H*256] bf16.
// Double-buffered K/VT tiles staged with global_load_async_to_lds_b128.
#define LK 264   // padded LDS stride for K tile (elems)
#define LV 40    // padded LDS stride for V^T tile
#define LP 40    // padded LDS stride for P scratch

__global__ __launch_bounds__(256)
void attn_kernel(const u16* __restrict__ qb, const u16* __restrict__ kb,
                 const u16* __restrict__ vtb, const float* __restrict__ mask,
                 u16* __restrict__ ob) {
  __shared__ u16 ldsK[2 * 32 * LK];    // K tiles: [buf][key 0..31][d 0..255]
  __shared__ u16 ldsVT[2 * 256 * LV];  // V^T tiles: [buf][d 0..255][key 0..31]
  __shared__ u16 ldsP[8 * 16 * LP];    // per-wave P scratch [16][32]

  const int lane = threadIdx.x & 31;
  const int wave = threadIdx.x >> 5;
  const int hf = lane >> 4, r = lane & 15;

  const int bh = blockIdx.y;  // b*H + h
  const int b = bh >> 3;
  const int h = bh & 7;
  const int q0 = blockIdx.x * 128 + wave * 16;

  const u16* qbase  = qb  + (size_t)b * 2048 * 2048 + h * 256;
  const u16* kbase  = kb  + (size_t)b * 2048 * 256;
  const u16* vtbase = vtb + (size_t)b * 256 * 2048;
  const float* mbase = mask + (size_t)b * 2048 * 2048;

  const u32 ldsK_addr  = (u32)(uintptr_t)(void*)ldsK;
  const u32 ldsVT_addr = (u32)(uintptr_t)(void*)ldsVT;

  // issue 8 async b128 copies (per thread) for one 32-key tile into buffer bi
  auto issue_tile = [&](int jt, int bi) {
    const int j0 = jt * 32;
    const int t = threadIdx.x;
    {   // K tile: 32 rows x 256 d, row-major contiguous
      const int row = t >> 3;
      const int cc  = (t & 7) * 32;
      const u32 lb = ldsK_addr + (u32)(bi * (32 * LK) + row * LK + cc) * 2;
      const u16* g = kbase + (size_t)(j0 + row) * 256 + cc;
#pragma unroll
      for (int i = 0; i < 4; ++i) async_b128(lb + i * 16, g + i * 8);
    }
    {   // V^T tile: 256 rows (d) x 32 keys, row contiguous in vtb
      const int d = t;
      const u32 lb = ldsVT_addr + (u32)(bi * (256 * LV) + d * LV) * 2;
      const u16* g = vtbase + (size_t)d * 2048 + j0;
#pragma unroll
      for (int i = 0; i < 4; ++i) async_b128(lb + i * 16, g + i * 8);
    }
  };

  // Preload this wave's 16x256 Q block as 8 A-fragments.
  v16bf qfrag[8];
#pragma unroll
  for (int d = 0; d < 8; ++d) qfrag[d] = load_a(qbase, 2048, q0, d * 32, lane);

  v8f o[16] = {};               // O accumulator: 16 D-tiles of 16x16
  float mrow[8], lrow[8];
#pragma unroll
  for (int g = 0; g < 8; ++g) { mrow[g] = -1e30f; lrow[g] = 0.f; }

  u16* myP = ldsP + wave * 16 * LP;

  issue_tile(0, 0);                      // prefetch first tile

  for (int jt = 0; jt < 64; ++jt) {      // 64 tiles of 32 keys
    const int j0 = jt * 32;
    if (jt + 1 < 64) {
      issue_tile(jt + 1, (jt + 1) & 1);  // prefetch next tile into other buffer
      asm volatile("s_wait_asynccnt 0x8" ::: "memory");  // current tile's 8 done
    } else {
      asm volatile("s_wait_asynccnt 0x0" ::: "memory");
    }
    __syncthreads();

    const u16* curK  = ldsK  + (jt & 1) * (32 * LK);
    const u16* curVT = ldsVT + (jt & 1) * (256 * LV);

    // S = Q * K^T for 32 keys (two 16x16 C tiles)
    v8f s0 = {}, s1 = {};
#pragma unroll
    for (int d = 0; d < 8; ++d) {
      v16bf kf0 = load_b(curK, LK, 0,  d * 32, lane);
      v16bf kf1 = load_b(curK, LK, 16, d * 32, lane);
      s0 = wmma_bf16(qfrag[d], kf0, s0);
      s1 = wmma_bf16(qfrag[d], kf1, s1);
    }

    // online softmax (rows live across the 16 lanes of each half-wave)
    float p0[8], p1[8];
#pragma unroll
    for (int g = 0; g < 8; ++g) {
      const int qr = q0 + g + 8 * hf;
      float sc0 = s0[g] * 0.0625f + mbase[(size_t)qr * 2048 + j0 + r];
      float sc1 = s1[g] * 0.0625f + mbase[(size_t)qr * 2048 + j0 + 16 + r];
      float v = fmaxf(sc0, sc1);
#pragma unroll
      for (int off = 8; off; off >>= 1) v = fmaxf(v, __shfl_xor(v, off, 32));
      const float mn = fmaxf(mrow[g], v);
      const float alpha = __expf(mrow[g] - mn);
      const float e0 = __expf(sc0 - mn);
      const float e1 = __expf(sc1 - mn);
      float rs = e0 + e1;
#pragma unroll
      for (int off = 8; off; off >>= 1) rs += __shfl_xor(rs, off, 32);
      lrow[g] = lrow[g] * alpha + rs;
      mrow[g] = mn;
      p0[g] = e0; p1[g] = e1;
#pragma unroll
      for (int t = 0; t < 16; ++t) o[t][g] *= alpha;
    }

    // C-layout P -> A-layout via per-wave LDS scratch
#pragma unroll
    for (int g = 0; g < 8; ++g) {
      myP[(g + 8 * hf) * LP + r]      = f2bf(p0[g]);
      myP[(g + 8 * hf) * LP + 16 + r] = f2bf(p1[g]);
    }
    asm volatile("s_wait_dscnt 0x0" ::: "memory");  // wave-local LDS RAW fence
    v16bf pf = load_a(myP, LP, 0, 0, lane);

    // O += P * V   (16 D-tiles)
#pragma unroll
    for (int t = 0; t < 16; ++t) {
      v16bf vf = load_b(curVT, LV, t * 16, 0, lane);
      o[t] = wmma_bf16(pf, vf, o[t]);
    }
    __syncthreads();   // all waves done with this buffer before it is refilled
  }

  // normalize and store to [B, L, H*256] bf16
  u16* obase = ob + (size_t)b * 2048 * 2048 + h * 256;
#pragma unroll
  for (int t = 0; t < 16; ++t) {
#pragma unroll
    for (int g = 0; g < 8; ++g) {
      const int qr = q0 + g + 8 * hf;
      obase[(size_t)qr * 2048 + t * 16 + r] = f2bf(o[t][g] / lrow[g]);
    }
  }
}

// ---------------------------------------------------------------- launch
extern "C" void kernel_launch(void* const* d_in, const int* in_sizes, int n_in,
                              void* d_out, int out_size, void* d_ws, size_t ws_size,
                              hipStream_t stream) {
  const float* hidden = (const float*)d_in[0];
  const float* mask   = (const float*)d_in[1];
  const float* cosb   = (const float*)d_in[2];
  const float* sinb   = (const float*)d_in[3];
  const float* Wq     = (const float*)d_in[4];
  const float* Wk     = (const float*)d_in[5];
  const float* Wv     = (const float*)d_in[6];
  const float* Wo     = (const float*)d_in[7];
  float* out = (float*)d_out;

  const int Bc = 2, Lc = 2048, Hc = 8, Dc = 256, Ec = 2048;
  const size_t ML = (size_t)Bc * Lc;  // 4096 token rows

  char* ws = (char*)d_ws;
  u16* hb   = (u16*)ws; ws += ML * Ec * 2;                 // hidden bf16      16 MB
  u16* wqb  = (u16*)ws; ws += (size_t)Hc * Dc * Ec * 2;    // Wq bf16           8 MB
  u16* wkb  = (u16*)ws; ws += (size_t)Dc * Ec * 2;         // Wk bf16           1 MB
  u16* wvb  = (u16*)ws; ws += (size_t)Dc * Ec * 2;         // Wv bf16           1 MB
  u16* wob  = (u16*)ws; ws += (size_t)Ec * Hc * Dc * 2;    // Wo bf16           8 MB
  u16* qbuf = (u16*)ws; ws += ML * Hc * Dc * 2;            // Q bf16           16 MB
  u16* kbuf = (u16*)ws; ws += ML * Dc * 2;                 // K bf16            2 MB
  u16* vbuf = (u16*)ws; ws += ML * Dc * 2;                 // V bf16            2 MB
  u16* vtb  = (u16*)ws; ws += ML * Dc * 2;                 // V^T bf16          2 MB
  u16* obuf = (u16*)ws; ws += ML * Hc * Dc * 2;            // attn out bf16    16 MB

  cvt_kernel<<<dim3(2048), 256, 0, stream>>>(hidden, hb, ML * Ec);
  cvt_kernel<<<dim3(2048), 256, 0, stream>>>(Wq, wqb, (size_t)Hc * Dc * Ec);
  cvt_kernel<<<dim3(512),  256, 0, stream>>>(Wk, wkb, (size_t)Dc * Ec);
  cvt_kernel<<<dim3(512),  256, 0, stream>>>(Wv, wvb, (size_t)Dc * Ec);
  cvt_kernel<<<dim3(2048), 256, 0, stream>>>(Wo, wob, (size_t)Ec * Hc * Dc);

  // projections: q = h*Wq^T, k = h*Wk^T, v = h*Wv^T
  gemm_kernel<true><<<dim3(Hc * Dc / 128, ML / 64), 256, 0, stream>>>(hb, wqb, qbuf, (int)ML, Hc * Dc, Ec);
  gemm_kernel<true><<<dim3(Dc / 128,      ML / 64), 256, 0, stream>>>(hb, wkb, kbuf, (int)ML, Dc, Ec);
  gemm_kernel<true><<<dim3(Dc / 128,      ML / 64), 256, 0, stream>>>(hb, wvb, vbuf, (int)ML, Dc, Ec);

  // RoPE in place
  rope_kernel<<<dim3(4096), 256, 0, stream>>>(qbuf, cosb, sinb, Lc, Hc, (size_t)Bc * Lc * Hc * 128);
  rope_kernel<<<dim3(1024), 256, 0, stream>>>(kbuf, cosb, sinb, Lc, 1,  (size_t)Bc * Lc * 128);

  // V -> V^T for async row-contiguous tile staging
  transpose_v_kernel<<<dim3(1024), 256, 0, stream>>>(vbuf, vtb, (size_t)Bc * Dc * Lc);

  // fused flash attention (double-buffered async LDS staging)
  attn_kernel<<<dim3(Lc / 128, Bc * Hc), 256, 0, stream>>>(qbuf, kbuf, vtb, mask, obuf);

  // output projection: out = attn_out * Wo^T  (f32 result)
  gemm_kernel<false><<<dim3(Ec / 128, ML / 64), 256, 0, stream>>>(obuf, wob, out, (int)ML, Ec, Hc * Dc);
}
